// CorrelationCost_14139032339205
// MI455X (gfx1250) — matrix operand order repaired and checked by reading, not compiled, not run on hardware
//
#include <hip/hip_runtime.h>

// CDNA5 (gfx1250) wave32 WMMA types
typedef __attribute__((ext_vector_type(2)))  float  v2f;
typedef __attribute__((ext_vector_type(8)))  float  v8f;
typedef __attribute__((ext_vector_type(8)))  __bf16 v8bf;
typedef __attribute__((ext_vector_type(16))) __bf16 v16bf;

#define CC_B  16
#define CC_H  48
#define CC_W  64
#define CC_C  256
#define CC_D  21            // displacements per axis (MD=20, S2=2)
#define CC_OD (CC_D * CC_D) // 441

// ---------------------------------------------------------------------------
// f32 -> bf16 round-to-nearest-even on the raw bits
// ---------------------------------------------------------------------------
__device__ __forceinline__ unsigned short bf16_rne(unsigned int u) {
  return (unsigned short)((u + 0x7FFFu + ((u >> 16) & 1u)) >> 16);
}

// ---------------------------------------------------------------------------
// Kernel 0: split f32 tensor into bf16 hi/lo planes (x ~= hi + lo, err<=2^-18)
// 8 elements per thread -> float4 x2 in, 16B hi + 16B lo out.
// ---------------------------------------------------------------------------
__global__ __launch_bounds__(256) void cc_split_bf16(const float* __restrict__ src,
                                                     unsigned short* __restrict__ hi,
                                                     unsigned short* __restrict__ lo,
                                                     int n8) {
  int i = blockIdx.x * blockDim.x + threadIdx.x;
  if (i >= n8) return;
  const float4* s4 = reinterpret_cast<const float4*>(src) + 2 * (size_t)i;
  float x[8];
  float4 t0 = s4[0], t1 = s4[1];
  x[0] = t0.x; x[1] = t0.y; x[2] = t0.z; x[3] = t0.w;
  x[4] = t1.x; x[5] = t1.y; x[6] = t1.z; x[7] = t1.w;
  union { unsigned short us[8]; uint4 u4; } H, L;
#pragma unroll
  for (int j = 0; j < 8; ++j) {
    unsigned int u  = __float_as_uint(x[j]);
    unsigned short h = bf16_rne(u);
    float r = x[j] - __uint_as_float((unsigned int)h << 16);
    H.us[j] = h;
    L.us[j] = bf16_rne(__float_as_uint(r));
  }
  reinterpret_cast<uint4*>(hi)[i] = H.u4;
  reinterpret_cast<uint4*>(lo)[i] = L.u4;
}

// ---------------------------------------------------------------------------
// Kernel 1: zero-fill the output (covers OOB dy slabs and clipped dx entries).
// ---------------------------------------------------------------------------
__global__ __launch_bounds__(256) void cc_fill_zero(float* __restrict__ out, int n) {
  const int n4 = n >> 2;
  float4 z = make_float4(0.f, 0.f, 0.f, 0.f);
  for (int i = blockIdx.x * blockDim.x + threadIdx.x; i < n4;
       i += gridDim.x * blockDim.x) {
    reinterpret_cast<float4*>(out)[i] = z;
  }
  for (int i = (n4 << 2) + blockIdx.x * blockDim.x + threadIdx.x; i < n;
       i += gridDim.x * blockDim.x) {
    out[i] = 0.f;
  }
}

// ---------------------------------------------------------------------------
// Kernel 2 (main): banded Gram correlation on the bf16 WMMA pipe, split-
// precision (hi*hi + hi*lo + lo*hi) accumulated in f32 -> f32-faithful.
//
// For fixed (b,h,dy): G[w,w'] = sum_c a[h,w,c]*b[h+dy,w',c];
// out[w, ti] = G[w, w + 2*(ti-10)] / 256. Rows split by w-parity so every
// Gram entry of a tile has even dx. 8 tile pairs per (b,h,dy) = 8 waves.
// K=256 in chunks of 32 via v_wmma_f32_16x16x32_bf16 (3 WMMAs/chunk).
// ---------------------------------------------------------------------------
__global__ __launch_bounds__(256) void cc_corr_wmma_bf16(
    const unsigned short* __restrict__ aHi, const unsigned short* __restrict__ aLo,
    const unsigned short* __restrict__ bHi, const unsigned short* __restrict__ bLo,
    float* __restrict__ out) {
  const int tj = blockIdx.x;          // dy index, 0..20
  const int bh = blockIdx.y;          // b*H + h
  const int b  = bh / CC_H;
  const int h  = bh - b * CC_H;
  const int h2 = h + (tj * 2 - 20);
  if (h2 < 0 || h2 >= CC_H) return;   // block-uniform: EXEC stays all-1s

  const int lane = threadIdx.x & 31;
  const int tp   = threadIdx.x >> 5;  // tile-pair id, 0..7 (one per wave)
  const int par  = tp & 1;            // w parity
  const int q    = tp >> 1;
  const int w0A  = (q >> 1) * 32;     // 0 or 32
  const int w0B  = (q & 1)  * 32;     // 0 or 32

  const int r  = lane & 15;           // M (A-tile row) / N (B-tile column)
  const int hi = lane >> 4;           // lane-half selects K sub-ranges

  const size_t rowA = (((size_t)bh * CC_W) + (size_t)(w0A + 2 * r + par)) * CC_C;
  const size_t rowB =
      ((((size_t)b * CC_H + h2) * CC_W) + (size_t)(w0B + 2 * r + par)) * CC_C;

  const __bf16* aH = reinterpret_cast<const __bf16*>(aHi) + rowA;
  const __bf16* aL = reinterpret_cast<const __bf16*>(aLo) + rowA;
  const __bf16* bH = reinterpret_cast<const __bf16*>(bHi) + rowB;
  const __bf16* bL = reinterpret_cast<const __bf16*>(bLo) + rowB;

  v8f acc = {};
#pragma unroll 2
  for (int k = 0; k < CC_C; k += 32) {
    // A 16x32 bf16 layout: lane<16 -> K[0..7]||K[16..23]; lane>=16 -> +8
    const int ca = k + 8 * hi;
    v8bf a0h = *reinterpret_cast<const v8bf*>(aH + ca);
    v8bf a1h = *reinterpret_cast<const v8bf*>(aH + ca + 16);
    v8bf a0l = *reinterpret_cast<const v8bf*>(aL + ca);
    v8bf a1l = *reinterpret_cast<const v8bf*>(aL + ca + 16);
    v16bf Ah = __builtin_shufflevector(a0h, a1h, 0,1,2,3,4,5,6,7,8,9,10,11,12,13,14,15);
    v16bf Al = __builtin_shufflevector(a0l, a1l, 0,1,2,3,4,5,6,7,8,9,10,11,12,13,14,15);
    // B 32x16 bf16 layout: lane<16 -> K[0..7]||K[16..23] of its column; lane>=16 -> +8
    const int cb = k + 8 * hi;
    v8bf b0h = *reinterpret_cast<const v8bf*>(bH + cb);
    v8bf b1h = *reinterpret_cast<const v8bf*>(bH + cb + 16);
    v8bf b0l = *reinterpret_cast<const v8bf*>(bL + cb);
    v8bf b1l = *reinterpret_cast<const v8bf*>(bL + cb + 16);
    v16bf Bh = __builtin_shufflevector(b0h, b1h, 0,1,2,3,4,5,6,7,8,9,10,11,12,13,14,15);
    v16bf Bl = __builtin_shufflevector(b0l, b1l, 0,1,2,3,4,5,6,7,8,9,10,11,12,13,14,15);

    acc = __builtin_amdgcn_wmma_f32_16x16x32_bf16(false, Ah, false, Bh,
                                                  (short)0, acc, false, false);
    acc = __builtin_amdgcn_wmma_f32_16x16x32_bf16(false, Ah, false, Bl,
                                                  (short)0, acc, false, false);
    acc = __builtin_amdgcn_wmma_f32_16x16x32_bf16(false, Al, false, Bh,
                                                  (short)0, acc, false, false);
  }

  // C/D layout: element v of lane holds G[M,N], M = v + 8*hi, N = lane&15.
  const float scale = 1.0f / (float)CC_C;
  const int dBase = (w0B - w0A) >> 1;
  float* outBase = out + ((size_t)bh * CC_W) * CC_OD + tj * CC_D;
#pragma unroll
  for (int v = 0; v < 8; ++v) {
    const int m  = v + 8 * hi;
    const int ti = dBase + r - m + 10;           // dx = 2*(ti-10)
    if (ti >= 0 && ti < CC_D) {
      const int w = w0A + 2 * m + par;
      outBase[(size_t)w * CC_OD + ti] = acc[v] * scale;
    }
  }
}

// ---------------------------------------------------------------------------
// Fallback (exact f32 pipe): v_wmma_f32_16x16x4_f32, used if ws is too small.
// ---------------------------------------------------------------------------
__global__ __launch_bounds__(256) void cc_corr_wmma_f32(const float* __restrict__ A,
                                                        const float* __restrict__ Bm,
                                                        float* __restrict__ out) {
  const int tj = blockIdx.x;
  const int bh = blockIdx.y;
  const int b  = bh / CC_H;
  const int h  = bh - b * CC_H;
  const int h2 = h + (tj * 2 - 20);
  if (h2 < 0 || h2 >= CC_H) return;

  const int lane = threadIdx.x & 31;
  const int tp   = threadIdx.x >> 5;
  const int par  = tp & 1;
  const int q    = tp >> 1;
  const int w0A  = (q >> 1) * 32;
  const int w0B  = (q & 1)  * 32;
  const int r    = lane & 15;
  const int hi   = lane >> 4;

  const float* aPtr =
      A + (((size_t)bh * CC_W) + (size_t)(w0A + 2 * r + par)) * CC_C + 2 * hi;
  const float* bPtr =
      Bm + ((((size_t)b * CC_H + h2) * CC_W) + (size_t)(w0B + 2 * r + par)) * CC_C +
      2 * hi;

  v8f acc = {};
#pragma unroll 4
  for (int k = 0; k < CC_C; k += 4) {
    v2f av = *reinterpret_cast<const v2f*>(aPtr + k);
    v2f bv = *reinterpret_cast<const v2f*>(bPtr + k);
    acc = __builtin_amdgcn_wmma_f32_16x16x4_f32(false, av, false, bv,
                                                (short)0, acc, false, false);
  }

  const float scale = 1.0f / (float)CC_C;
  const int dBase = (w0B - w0A) >> 1;
  float* outBase = out + ((size_t)bh * CC_W) * CC_OD + tj * CC_D;
#pragma unroll
  for (int v = 0; v < 8; ++v) {
    const int m  = v + 8 * hi;
    const int ti = dBase + r - m + 10;
    if (ti >= 0 && ti < CC_D) {
      const int w = w0A + 2 * m + par;
      outBase[(size_t)w * CC_OD + ti] = acc[v] * scale;
    }
  }
}

// ---------------------------------------------------------------------------
extern "C" void kernel_launch(void* const* d_in, const int* in_sizes, int n_in,
                              void* d_out, int out_size, void* d_ws, size_t ws_size,
                              hipStream_t stream) {
  (void)in_sizes; (void)n_in;
  const float* a  = (const float*)d_in[0];
  const float* bm = (const float*)d_in[1];
  float* out = (float*)d_out;

  // 1) zero-fill output (16*48*64*441 = 21,676,032 floats)
  {
    int n4 = out_size >> 2;
    int blocks = (n4 + 255) / 256;
    if (blocks < 1) blocks = 1;
    cc_fill_zero<<<blocks, 256, 0, stream>>>(out, out_size);
  }

  const size_t NA   = (size_t)CC_B * CC_H * CC_W * CC_C;  // 12,582,912 elements
  const size_t need = NA * 4 * sizeof(unsigned short);    // aHi,aLo,bHi,bLo

  if (ws_size >= need) {
    // 2) split both inputs into bf16 hi/lo planes
    unsigned short* aHi = (unsigned short*)d_ws;
    unsigned short* aLo = aHi + NA;
    unsigned short* bHi = aLo + NA;
    unsigned short* bLo = bHi + NA;
    const int n8 = (int)(NA / 8);
    const int cblocks = (n8 + 255) / 256;
    cc_split_bf16<<<cblocks, 256, 0, stream>>>(a,  aHi, aLo, n8);
    cc_split_bf16<<<cblocks, 256, 0, stream>>>(bm, bHi, bLo, n8);

    // 3) banded Gram correlation on the bf16 matrix pipe (split-precision)
    cc_corr_wmma_bf16<<<dim3(CC_D, CC_B * CC_H), 256, 0, stream>>>(aHi, aLo, bHi,
                                                                   bLo, out);
  } else {
    // Fallback: exact f32 WMMA pipe, no workspace needed
    cc_corr_wmma_f32<<<dim3(CC_D, CC_B * CC_H), 256, 0, stream>>>(a, bm, out);
  }
}
